// carry_save_adder_13786845020439
// MI455X (gfx1250) — matrix operand order repaired and checked by reading, not compiled, not run on hardware
//
#include <hip/hip_runtime.h>

typedef __attribute__((ext_vector_type(16))) _Float16 v16h;
typedef __attribute__((ext_vector_type(8)))  float    v8f;
typedef __attribute__((ext_vector_type(4)))  float    v4f;

// One wave32 per 16-row tile. K = 1024 (64 i-slots x 16 bits), reduced in 32
// chunks of K=32 via v_wmma_f32_16x16x32_f16. B fragment = bit weights 2^j
// (period 16 in K matches the half-j <-> K (mod 16) mapping this code uses for the
// 16-bit A/B operands, so it is one constant fragment for the whole loop).
__global__ __launch_bounds__(256) void csa_wmma_kernel(
    const float* __restrict__ x, const int* __restrict__ mask,
    float* __restrict__ out_bits, float* __restrict__ out_carry, int rows)
{
  const int lane = threadIdx.x & 31;
  const int wave = threadIdx.x >> 5;
  const int wavesPerBlock = blockDim.x >> 5;
  const int tile = blockIdx.x * wavesPerBlock + wave;
  const int row_base = tile * 16;
  if (row_base >= rows) return;            // uniform per wave

  const int half = lane >> 4;              // 0: lanes 0-15, 1: lanes 16-31
  const int m    = lane & 15;              // A-matrix row carried by this lane
  const float* __restrict__ xrow = x + (size_t)(row_base + m) * 1024;

  // A 16x32 in this code's contiguous k order: lanes 0-15 hold K={0..15}, lanes 16-31 K={16..31}
  const int kb0 = half ? 16 : 0;
  const int kb1 = kb0 + 8;

  // B fragment: b[j] = 2^j, identical in every lane / every chunk.
  v16h b;
  #pragma unroll
  for (int j = 0; j < 16; ++j) b[j] = (_Float16)(float)(1u << j);

  v8f acc = {};

  #pragma unroll 4
  for (int c = 0; c < 32; ++c) {
    const int koff = c * 32;
    // This lane-half's K range covers input slot i = 2*c + half (16 bits each).
    const int i = 2 * c + half;
    const bool keep = (i < 2) || (mask[i] != 0);

    v4f f0 = *(const v4f*)(xrow + koff + kb0);
    v4f f1 = *(const v4f*)(xrow + koff + kb0 + 4);
    v4f f2 = *(const v4f*)(xrow + koff + kb1);
    v4f f3 = *(const v4f*)(xrow + koff + kb1 + 4);

    v16h a;
    #pragma unroll
    for (int j = 0; j < 4; ++j) {
      a[j]      = (_Float16)f0[j];
      a[4 + j]  = (_Float16)f1[j];
      a[8 + j]  = (_Float16)f2[j];
      a[12 + j] = (_Float16)f3[j];
    }
    if (!keep) { v16h z = {}; a = z; }     // data-level masking; EXEC stays full

    // 8 args: (neg_a, A, neg_b, B, c_mod, C, reuse_a, reuse_b)
    acc = __builtin_amdgcn_wmma_f32_16x16x32_f16(
        false, a, false, b, (short)0, acc, false, false);
  }

  // C/D f32 16x16 layout: acc[r] = row (row_base + 8*half + r) at this lane.
  // Lane r writes rows 0..7, lane 16+r writes rows 8..15.
  #pragma unroll
  for (int r = 0; r < 8; ++r) {
    if (m == r) {
      const int row = row_base + half * 8 + r;
      const int t = (int)(acc[r] + 0.5f);  // exact integer < 2^22
      #pragma unroll
      for (int j = 0; j < 16; ++j)
        out_bits[(size_t)row * 16 + j] = (float)((t >> j) & 1);
      out_carry[row] = (float)(t >> 16);
    }
  }
}

extern "C" void kernel_launch(void* const* d_in, const int* in_sizes, int n_in,
                              void* d_out, int out_size, void* d_ws, size_t ws_size,
                              hipStream_t stream) {
  const float* x    = (const float*)d_in[0];
  const int*   mask = (const int*)d_in[1];
  float*       out  = (float*)d_out;

  const int rows = in_sizes[0] / (64 * 16);       // 32768
  float* out_bits  = out;                          // (rows, 16) flat
  float* out_carry = out + (size_t)rows * 16;      // (rows,)

  const int tiles = (rows + 15) / 16;              // one wave per tile
  const int wavesPerBlock = 8;                     // 256 threads
  const int blocks = (tiles + wavesPerBlock - 1) / wavesPerBlock;

  csa_wmma_kernel<<<blocks, wavesPerBlock * 32, 0, stream>>>(
      x, mask, out_bits, out_carry, rows);
}
